// MLPPredictor_76605036692176
// MI455X (gfx1250) — matrix operand order repaired and checked by reading, not compiled, not run on hardware
//
#include <hip/hip_runtime.h>
#include <cstdint>

// Problem constants (from reference)
#define NU 100000   // users
#define NM 50000    // movies
#define DD 128      // feature dim
#define EE 1000000  // edges
#define CC 5        // classes

typedef __attribute__((ext_vector_type(16))) __bf16    v16bf;
typedef __attribute__((ext_vector_type(2)))  __bf16    v2bf;
typedef __attribute__((ext_vector_type(8)))  float     v8f;
typedef __attribute__((ext_vector_type(8)))  uint32_t  v8u;

__device__ __forceinline__ v8f wmma_bf16(v16bf a, v16bf b, v8f c) {
    // D = A(16x32 bf16) * B(32x16 bf16) + C(16x16 f32)
    return __builtin_amdgcn_wmma_f32_16x16x32_bf16(false, a, false, b, (short)0, c, false, false);
}

// bf16 conversion via native casts (lowers to v_cvt_pk_bf16_f32, RNE)
__device__ __forceinline__ unsigned short bf16_bits(float f) {
    return __builtin_bit_cast(unsigned short, (__bf16)f);
}

// 8 f32 (a,b) -> frag elements 0..7 ; 8 f32 (c,d) -> elements 8..15
__device__ __forceinline__ v16bf pack8f_pair(float4 a, float4 b, float4 c, float4 d) {
    v16bf t;
    t[0]  = (__bf16)a.x; t[1]  = (__bf16)a.y; t[2]  = (__bf16)a.z; t[3]  = (__bf16)a.w;
    t[4]  = (__bf16)b.x; t[5]  = (__bf16)b.y; t[6]  = (__bf16)b.z; t[7]  = (__bf16)b.w;
    t[8]  = (__bf16)c.x; t[9]  = (__bf16)c.y; t[10] = (__bf16)c.z; t[11] = (__bf16)c.w;
    t[12] = (__bf16)d.x; t[13] = (__bf16)d.y; t[14] = (__bf16)d.z; t[15] = (__bf16)d.w;
    return t;
}

// B fragment of W2 padded from [5,128] to [16,128] (fused fallback only)
__device__ __forceinline__ v16bf w2_frag(const float* __restrict__ W2, int l16, int half, int ks) {
    if (l16 < CC) {
        const float4* q = (const float4*)(W2 + l16 * DD + ks * 32 + half * 16);
        return pack8f_pair(q[0], q[1], q[2], q[3]);
    }
    return (v16bf)((__bf16)0.0f);  // splat zero
}

// packed bf16 pair: relu(u + v) -> v_pk_add_bf16 + v_pk_max_num_bf16
__device__ __forceinline__ uint32_t comb_relu(uint32_t up, uint32_t vp) {
    v2bf a = __builtin_bit_cast(v2bf, up);
    v2bf b = __builtin_bit_cast(v2bf, vp);
    v2bf s = a + b;
    v2bf z = {(__bf16)0.0f, (__bf16)0.0f};
    v2bf r = __builtin_elementwise_max(s, z);
    return __builtin_bit_cast(uint32_t, r);
}

// =====================================================================
// Kernel 0: zero-padded bf16 copy of W2 -> W2p[16][128] (4 KB, L2-hot).
// Single block; rows 5..15 are zeros so the edge kernel's B path has
// no divergence and no conversions.
// =====================================================================
__global__ __launch_bounds__(256) void pad_w2_bf16(
    const float* __restrict__ W2, uint32_t* __restrict__ W2p)
{
    for (int idx = threadIdx.x; idx < 16 * 64; idx += 256) {
        int n = idx >> 6, j = idx & 63;  // row n, bf16-pair j
        uint32_t v = 0u;
        if (n < CC) {
            const float2* wr = (const float2*)(W2 + n * DD);
            float2 f = wr[j];
            v = (uint32_t)bf16_bits(f.x) | ((uint32_t)bf16_bits(f.y) << 16);
        }
        W2p[idx] = v;
    }
}

// =====================================================================
// Kernel 1: out_bf16[r, o] = sum_k feat[r,k] * W1[o, col_off + k]
// (precompute of U = ufeat @ W1[:, :128]^T and V = ifeat @ W1[:, 128:]^T)
// 256 threads = 8 waves; each wave computes a 16-row x 128-col tile.
// W1 half staged in LDS as bf16, pitch 68 u32 (136 bf16) -> conflict-free
// ds_load_b128 reads (bank stride 4 per lane).
// =====================================================================
__global__ __launch_bounds__(256) void feat_w1_gemm(
    const float* __restrict__ feat, int rows,
    const float* __restrict__ W1, int col_off,
    unsigned short* __restrict__ out)
{
    __shared__ uint32_t w1s[128 * 68];  // [o][k-pair], 64 pairs used + 4 pad

    const int tid = threadIdx.x;
    // Stage W1[:, col_off : col_off+128] as packed bf16 pairs
    for (int idx = tid; idx < 128 * 64; idx += 256) {
        int o = idx >> 6, j = idx & 63;
        const float2* wr = (const float2*)(W1 + o * 256 + col_off);
        float2 f = wr[j];
        w1s[o * 68 + j] = (uint32_t)bf16_bits(f.x) | ((uint32_t)bf16_bits(f.y) << 16);
    }
    __syncthreads();

    const int wv = tid >> 5, lane = tid & 31;
    const int half = lane >> 4, l16 = lane & 15;
    const int rbase = (blockIdx.x * 8 + wv) * 16;
    const bool full = (rbase + 16 <= rows);  // wave-uniform tail check

    // A fragments: 16 rows x K=128 (4 ksteps of 32), rows clamped for tails
    int r = rbase + l16; if (r > rows - 1) r = rows - 1;
    const float4* frow = (const float4*)(feat + (size_t)r * DD);
    v16bf A[4];
#pragma unroll
    for (int ks = 0; ks < 4; ++ks) {
        int c1 = ks * 32 + half * 16;  // f32 element offset
        A[ks] = pack8f_pair(frow[c1 / 4], frow[c1 / 4 + 1],
                            frow[c1 / 4 + 2], frow[c1 / 4 + 3]);
    }

#pragma unroll
    for (int nt = 0; nt < 8; ++nt) {
        v8f acc = {};
        const uint32_t* brow = &w1s[(nt * 16 + l16) * 68];
#pragma unroll
        for (int ks = 0; ks < 4; ++ks) {
            int j0 = (ks * 32 + half * 16) >> 1;  // pair index
            v8u t;
#pragma unroll
            for (int j = 0; j < 8; ++j) t[j] = brow[j0 + j];
            acc = wmma_bf16(A[ks], __builtin_bit_cast(v16bf, t), acc);
        }
        // Store D tile as bf16: element rr -> row rbase + rr + 8*half, col nt*16+l16
        unsigned short* op = out + (size_t)(rbase + half * 8) * DD + nt * 16 + l16;
        if (full) {
#pragma unroll
            for (int rr = 0; rr < 8; ++rr) op[(size_t)rr * DD] = bf16_bits(acc[rr]);
        } else {
#pragma unroll
            for (int rr = 0; rr < 8; ++rr)
                if (rbase + half * 8 + rr < rows) op[(size_t)rr * DD] = bf16_bits(acc[rr]);
        }
    }
}

// =====================================================================
// Kernel 2 (main path): per-edge  score = ReLU(U[src]+V[dst]) @ W2pad^T
// Each wave: 16 edges. Gathers bf16 rows from L2-resident U/V tables,
// fuses packed-bf16 add+ReLU into A fragments, loads pre-padded bf16 W2
// fragments (no cvt, no divergence), 4 bf16 WMMAs, stores 5 columns.
// =====================================================================
__global__ __launch_bounds__(256) void edge_mlp(
    const unsigned short* __restrict__ U, const unsigned short* __restrict__ V,
    const int* __restrict__ src, const int* __restrict__ dst,
    const unsigned short* __restrict__ W2p, float* __restrict__ out)
{
    const int tid = threadIdx.x;
    const int wv = tid >> 5, lane = tid & 31;
    const int half = lane >> 4, l16 = lane & 15;
    const int ebase = (blockIdx.x * 8 + wv) * 16;
    if (ebase >= EE) return;  // wave-uniform; no barriers in this kernel
    const bool full = (ebase + 16 <= EE);

    int e = ebase + l16; if (e > EE - 1) e = EE - 1;
    const int s = src[e], t = dst[e];
    const uint4* urow  = (const uint4*)(U + (size_t)s * DD);
    const uint4* vrow  = (const uint4*)(V + (size_t)t * DD);
    const uint4* w2row = (const uint4*)(W2p + l16 * DD);  // padded, rows>=5 zero

    v8f acc = {};
#pragma unroll
    for (int ks = 0; ks < 4; ++ks) {
        // A = ReLU(U[s]+V[d]): elements 0..15 at bf16 offset c1 = ks*32+half*16
        //  ->  uint4 indices ks*4+half*2 and ks*4+half*2+1
        int i0 = ks * 4 + half * 2;
        uint4 ua = urow[i0], ub = urow[i0 + 1];
        uint4 va = vrow[i0], vb = vrow[i0 + 1];
        v8u a;
        a[0] = comb_relu(ua.x, va.x); a[1] = comb_relu(ua.y, va.y);
        a[2] = comb_relu(ua.z, va.z); a[3] = comb_relu(ua.w, va.w);
        a[4] = comb_relu(ub.x, vb.x); a[5] = comb_relu(ub.y, vb.y);
        a[6] = comb_relu(ub.z, vb.z); a[7] = comb_relu(ub.w, vb.w);
        // B: 16 contiguous bf16 at koff = ks*32 + half*16 -> uint4 idx ks*4+half*2
        uint4 b0 = w2row[ks * 4 + half * 2], b1 = w2row[ks * 4 + half * 2 + 1];
        v8u b;
        b[0] = b0.x; b[1] = b0.y; b[2] = b0.z; b[3] = b0.w;
        b[4] = b1.x; b[5] = b1.y; b[6] = b1.z; b[7] = b1.w;
        acc = wmma_bf16(__builtin_bit_cast(v16bf, a), __builtin_bit_cast(v16bf, b), acc);
    }

    // D element rr -> edge ebase + rr + 8*half, class l16 (valid for l16 < 5)
    if (l16 < CC) {  // single divergent branch; stores unguarded on fast path
        float* op = out + (size_t)(ebase + half * 8) * CC + l16;
        if (full) {
#pragma unroll
            for (int rr = 0; rr < 8; ++rr) op[rr * CC] = acc[rr];
        } else {
#pragma unroll
            for (int rr = 0; rr < 8; ++rr)
                if (ebase + half * 8 + rr < EE) op[rr * CC] = acc[rr];
        }
    }
}

// =====================================================================
// Fallback (ws too small): fully fused per-edge 2-layer MLP on WMMA.
// W1 (both halves) staged in LDS as bf16; hidden tile transposed from
// the C/D layout into A-fragment layout through per-wave LDS staging.
// =====================================================================
__global__ __launch_bounds__(256) void edge_mlp_fused(
    const float* __restrict__ uf, const float* __restrict__ iff,
    const float* __restrict__ W1, const float* __restrict__ W2,
    const int* __restrict__ src, const int* __restrict__ dst,
    float* __restrict__ out)
{
    __shared__ uint32_t w1s[128 * 132];    // [o][k-pair], 128 pairs used + 4 pad
    __shared__ uint32_t hids[8 * 16 * 68]; // per-wave 16x128 bf16 hidden, pitch 68 u32

    const int tid = threadIdx.x;
    for (int idx = tid; idx < 128 * 128; idx += 256) {
        int o = idx >> 7, j = idx & 127;
        const float2* wr = (const float2*)(W1 + o * 256);
        float2 f = wr[j];
        w1s[o * 132 + j] = (uint32_t)bf16_bits(f.x) | ((uint32_t)bf16_bits(f.y) << 16);
    }
    __syncthreads();

    const int wv = tid >> 5, lane = tid & 31;
    const int half = lane >> 4, l16 = lane & 15;
    const int ebase = (blockIdx.x * 8 + wv) * 16;
    if (ebase >= EE) return;  // after barrier; wave-uniform
    const bool full = (ebase + 16 <= EE);

    int e = ebase + l16; if (e > EE - 1) e = EE - 1;
    const int s = src[e], t = dst[e];
    const float4* urow = (const float4*)(uf + (size_t)s * DD);
    const float4* vrow = (const float4*)(iff + (size_t)t * DD);

    // A1 = [h_u | h_v] as 16x32 bf16 fragments over K=256 (8 ksteps)
    v16bf A1[8];
#pragma unroll
    for (int ks = 0; ks < 8; ++ks) {
        const float4* rowp = (ks < 4) ? urow : vrow;
        int c1 = (ks & 3) * 32 + half * 16;
        A1[ks] = pack8f_pair(rowp[c1 / 4], rowp[c1 / 4 + 1],
                             rowp[c1 / 4 + 2], rowp[c1 / 4 + 3]);
    }

    // GEMM1 + ReLU -> LDS staging (bf16, 16 rows x 128 cols, pitch 136 bf16)
    unsigned short* hid16 = (unsigned short*)&hids[wv * 16 * 68];
#pragma unroll
    for (int nt = 0; nt < 8; ++nt) {
        v8f acc = {};
        const uint32_t* brow = &w1s[(nt * 16 + l16) * 132];
#pragma unroll
        for (int ks = 0; ks < 8; ++ks) {
            int j0 = (ks * 32 + half * 16) >> 1;
            v8u bt;
#pragma unroll
            for (int j = 0; j < 8; ++j) bt[j] = brow[j0 + j];
            acc = wmma_bf16(A1[ks], __builtin_bit_cast(v16bf, bt), acc);
        }
#pragma unroll
        for (int rr = 0; rr < 8; ++rr) {
            int m = rr + half * 8;  // row within tile
            hid16[m * 136 + nt * 16 + l16] = bf16_bits(fmaxf(acc[rr], 0.0f));
        }
    }
    // Same-wave DS ops are in order; compiler inserts s_wait_dscnt on reuse.

    // GEMM2: A2 from LDS hidden, B = padded W2
    const uint32_t* hrow = &hids[wv * 16 * 68 + l16 * 68];
    v8f acc = {};
#pragma unroll
    for (int ks = 0; ks < 4; ++ks) {
        int p0 = ks * 16 + half * 8;  // pair index of bf16 offset ks*32+half*16
        v8u a;
#pragma unroll
        for (int j = 0; j < 4; ++j) { a[j] = hrow[p0 + j]; a[4 + j] = hrow[p0 + 4 + j]; }
        acc = wmma_bf16(__builtin_bit_cast(v16bf, a), w2_frag(W2, l16, half, ks), acc);
    }

    if (l16 < CC) {
        float* op = out + (size_t)(ebase + half * 8) * CC + l16;
        if (full) {
#pragma unroll
            for (int rr = 0; rr < 8; ++rr) op[rr * CC] = acc[rr];
        } else {
#pragma unroll
            for (int rr = 0; rr < 8; ++rr)
                if (ebase + half * 8 + rr < EE) op[rr * CC] = acc[rr];
        }
    }
}

// =====================================================================
extern "C" void kernel_launch(void* const* d_in, const int* in_sizes, int n_in,
                              void* d_out, int out_size, void* d_ws, size_t ws_size,
                              hipStream_t stream) {
    const float* ufeat = (const float*)d_in[0];
    const float* ifeat = (const float*)d_in[1];
    const float* W1    = (const float*)d_in[2];
    const float* W2    = (const float*)d_in[3];
    const int*   src   = (const int*)d_in[4];
    const int*   dst   = (const int*)d_in[5];
    float* out = (float*)d_out;

    const size_t needU = (size_t)NU * DD * sizeof(unsigned short);  // 25.6 MB
    const size_t needV = (size_t)NM * DD * sizeof(unsigned short);  // 12.8 MB
    const size_t needW = (size_t)16 * DD * sizeof(unsigned short);  // 4 KB padded W2

    if (ws_size >= needU + needV + needW) {
        // Factored path: hidden = ReLU(U[src] + V[dst]); 50x fewer FLOPs,
        // bf16 tables halve the L2 gather traffic.
        unsigned short* U   = (unsigned short*)d_ws;
        unsigned short* V   = (unsigned short*)((char*)d_ws + needU);
        unsigned short* W2p = (unsigned short*)((char*)d_ws + needU + needV);
        pad_w2_bf16<<<1, 256, 0, stream>>>(W2, (uint32_t*)W2p);
        feat_w1_gemm<<<(NU + 127) / 128, 256, 0, stream>>>(ufeat, NU, W1, 0,   U);
        feat_w1_gemm<<<(NM + 127) / 128, 256, 0, stream>>>(ifeat, NM, W1, 128, V);
        edge_mlp<<<(EE + 127) / 128, 256, 0, stream>>>(U, V, src, dst, W2p, out);
    } else {
        edge_mlp_fused<<<(EE + 127) / 128, 256, 0, stream>>>(
            ufeat, ifeat, W1, W2, src, dst, out);
    }
}
